// UnifiedMultiLevel_22368189677911
// MI455X (gfx1250) — matrix operand order repaired and checked by reading, not compiled, not run on hardware
//
#include <hip/hip_runtime.h>
#include <hip/hip_bf16.h>
#include <math.h>

typedef _Float16 f16;
typedef __attribute__((ext_vector_type(16))) _Float16 v16h;
typedef __attribute__((ext_vector_type(8)))  _Float16 v8h;
typedef __attribute__((ext_vector_type(8)))  float    v8f;

// Problem constants: B=64, C=128, L=1024, LEVELS=3, M = B*C = 8192 GEMM rows.

__device__ __forceinline__ float gelu_exact(float x) {
    return 0.5f * x * (1.0f + erff(x * 0.70710678118654752440f));
}

__device__ __forceinline__ float wave_sum(float v) {
#pragma unroll
    for (int off = 16; off > 0; off >>= 1) v += __shfl_xor(v, off, 32);
    return v;
}

union FragU { v16h v; v8h h[2]; };

// ---------------------------------------------------------------------------
// Fused WMMA GEMM:  acc = A(8192xK) x W(KxK, stored row-major as (N,K)) + bias
// Block: 8 waves; wave w -> rows [blockIdx.x*128 + w*16, +16),
// cols [blockIdx.y*64, +64) as 4 WMMA accumulators. W strip staged in LDS.
// MODE 0: forward predict : d  = cur[m,2n+1] - gelu(acc+b); out32=d, out16=d,
//                           per-batch sum|d|
// MODE 1: forward update  : c  = cur[m,2n]   + gelu(acc+b); out32=c,
//                           per-batch sum|c|, global sum c^2
// MODE 2: recon update    : xe = approx[m,n] - gelu(acc+b); out32=xe, out16=xe
// MODE 3: recon predict   : xo = det[m,n]    + gelu(acc+b);
//                           out32 (row len 2K): [2n]=xe32[m,n], [2n+1]=xo
// ---------------------------------------------------------------------------
template<int MODE, int K>
__global__ void __launch_bounds__(256) gemm_fused(
    const f16* __restrict__ A, const f16* __restrict__ W,
    const float* __restrict__ bias,
    const float* __restrict__ aux,
    float* __restrict__ out32, f16* __restrict__ out16,
    const float* __restrict__ xe32,
    float* __restrict__ sumAbs, float* __restrict__ sumSq)
{
    constexpr int LDK = K + 8;            // pad 16B/row -> no 1KB-stride bank conflicts
    extern __shared__ char smem_raw[];
    f16* sW = (f16*)smem_raw;             // [64][LDK]

    const int tid    = threadIdx.x;
    const int lane   = tid & 31;
    const int wave   = tid >> 5;
    const int tileM  = blockIdx.x * 128 + wave * 16;
    const int tileN0 = blockIdx.y * 64;
    const int r15    = lane & 15;
    const int hi     = lane >> 4;

    // ---- cooperative stage of W[tileN0 .. tileN0+63][0..K) into LDS ----
    constexpr int CPR = K >> 3;           // 16B chunks per row
    constexpr int NCH = CPR * 64;
    for (int c = tid; c < NCH; c += 256) {
        int row = c / CPR;
        int col = (c - row * CPR) << 3;
        *(v8h*)(sW + row * LDK + col) =
            *(const v8h*)(W + (size_t)(tileN0 + row) * K + col);
    }
    __syncthreads();

    // A fragment: row m = tileM+r15; lanes 0-15 hold K {0..7,16..23},
    //             lanes 16-31 hold K {8..15,24..31}   (16-bit A layout)
    const f16* ap = A + (size_t)(tileM + r15) * K + hi * 8;
    // B fragment (from LDS): col n = local row r15 of strip;
    //             lanes 0-15 hold K {0..7,16..23}, lanes 16-31 hold K {8..15,24..31}
    const f16* bp = sW + r15 * LDK + hi * 8;

    v8f acc[4] = {};
#pragma unroll 4
    for (int kb = 0; kb < K; kb += 32) {
        FragU a;
        a.h[0] = *(const v8h*)(ap + kb);
        a.h[1] = *(const v8h*)(ap + kb + 16);
        if (kb + 32 < K) __builtin_prefetch(ap + kb + 32, 0, 1);

        // Pre-load all 4 B fragments so the 8 ds_load_b128 can clause up and
        // the 4 WMMAs issue back-to-back behind a single DScnt wait.
        FragU b[4];
#pragma unroll
        for (int t = 0; t < 4; ++t) {
            const f16* bt = bp + t * (16 * LDK) + kb;
            b[t].h[0] = *(const v8h*)(bt);
            b[t].h[1] = *(const v8h*)(bt + 16);
        }
#pragma unroll
        for (int t = 0; t < 4; ++t) {
            acc[t] = __builtin_amdgcn_wmma_f32_16x16x32_f16(
                         false, a.v, false, b[t].v, (short)0, acc[t], false, false);
        }
    }

    // ---- fused epilogue ----
    float pAbs = 0.f, pSq = 0.f;
#pragma unroll
    for (int t = 0; t < 4; ++t) {
        const int n = tileN0 + t * 16 + r15;
        const float bn = bias[n];
#pragma unroll
        for (int j = 0; j < 8; ++j) {
            const int m = tileM + j + hi * 8;         // C/D layout: VGPR j -> row
            const float gl = gelu_exact(acc[t][j] + bn);
            if (MODE == 0) {
                float r = aux[(size_t)m * 2 * K + 2 * n + 1] - gl;   // xo - gelu
                pAbs += fabsf(r);
                out32[(size_t)m * K + n] = r;
                out16[(size_t)m * K + n] = (f16)r;
            } else if (MODE == 1) {
                float r = aux[(size_t)m * 2 * K + 2 * n] + gl;       // xe + gelu
                pAbs += fabsf(r);
                pSq  += r * r;
                out32[(size_t)m * K + n] = r;
            } else if (MODE == 2) {
                float r = aux[(size_t)m * K + n] - gl;               // approx - gelu
                out32[(size_t)m * K + n] = r;
                out16[(size_t)m * K + n] = (f16)r;
            } else {
                float xo = aux[(size_t)m * K + n] + gl;              // det + gelu
                out32[(size_t)m * 2 * K + 2 * n]     = xe32[(size_t)m * K + n];
                out32[(size_t)m * 2 * K + 2 * n + 1] = xo;
            }
        }
    }

    if (MODE == 0 || MODE == 1) {
        // 128 rows per batch, 16-row tiles -> batch is uniform per wave
        float s = wave_sum(pAbs);
        if (MODE == 1) {
            float q = wave_sum(pSq);
            if (lane == 0) atomicAdd(sumSq, q);
        }
        if (lane == 0) atomicAdd(&sumAbs[tileM >> 7], s);
    }
}

template<int MODE>
static inline void launch_gemm(int D, hipStream_t stream,
                               const f16* A, const f16* W, const float* bias,
                               const float* aux, float* out32, f16* out16,
                               const float* xe32, float* sumAbs, float* sumSq)
{
    dim3 grid(64, D / 64);
    size_t sh = (size_t)64 * (D + 8) * sizeof(f16);
    switch (D) {
    case 512: gemm_fused<MODE, 512><<<grid, 256, sh, stream>>>(A, W, bias, aux, out32, out16, xe32, sumAbs, sumSq); break;
    case 256: gemm_fused<MODE, 256><<<grid, 256, sh, stream>>>(A, W, bias, aux, out32, out16, xe32, sumAbs, sumSq); break;
    default:  gemm_fused<MODE, 128><<<grid, 256, sh, stream>>>(A, W, bias, aux, out32, out16, xe32, sumAbs, sumSq); break;
    }
}

// ---------------------------------------------------------------------------
__global__ void f32_to_f16(const float* __restrict__ src, f16* __restrict__ dst, int n) {
    int i = blockIdx.x * blockDim.x + threadIdx.x;
    if (i < n) dst[i] = (f16)src[i];
}

// xe16[m,n] = (f16) cur[m, 2n];  accumulate sum(xe^2) for the level
__global__ void split_even(const float* __restrict__ cur, f16* __restrict__ xe16,
                           int D, size_t total, float* __restrict__ sumSqXe) {
    size_t stride = (size_t)gridDim.x * blockDim.x;
    float acc = 0.f;
    for (size_t i = (size_t)blockIdx.x * blockDim.x + threadIdx.x; i < total; i += stride) {
        size_t m = i / (size_t)D;
        int    nn = (int)(i - m * (size_t)D);
        float  v = cur[m * (size_t)(2 * D) + 2 * nn];
        xe16[i] = (f16)v;
        acc += v * v;
    }
    acc = wave_sum(acc);
    if ((threadIdx.x & 31) == 0) atomicAdd(sumSqXe, acc);
}

// gated = coeffs * g[b, gidx]; optional f16 copy for the recon GEMM A operand
__global__ void gate_scale(const float* __restrict__ coeffs, const float* __restrict__ g,
                           int gidx, float* __restrict__ gated, f16* __restrict__ det16,
                           size_t total, int perBatch) {
    size_t stride = (size_t)gridDim.x * blockDim.x;
    for (size_t i = (size_t)blockIdx.x * blockDim.x + threadIdx.x; i < total; i += stride) {
        int b = (int)(i / (size_t)perBatch);
        float v = coeffs[i] * g[b * 4 + gidx];
        gated[i] = v;
        if (det16) det16[i] = (f16)v;
    }
}

// energies (B,2) per level, gate sigmoid, scalar loss. One block, 64 threads.
__global__ void gate_and_loss(const float* __restrict__ sumAbsC,
                              const float* __restrict__ sumAbsD,
                              const float* __restrict__ sumSqXe,
                              const float* __restrict__ sumSqC,
                              const float* __restrict__ gate_w,
                              const float* __restrict__ gate_b,
                              float* __restrict__ g,
                              float* __restrict__ e0, float* __restrict__ e1,
                              float* __restrict__ e2, float* __restrict__ e3,
                              float* __restrict__ loss_out) {
    int b = threadIdx.x;
    if (b < 64) {
        const float n0 = 128.f * 512.f, n1 = 128.f * 256.f, n2 = 128.f * 128.f;
        float ce0 = sumAbsC[b]       / n0, de0 = sumAbsD[b]       / n0;
        float ce1 = sumAbsC[64 + b]  / n1, de1 = sumAbsD[64 + b]  / n1;
        float ce2 = sumAbsC[128 + b] / n2, de2 = sumAbsD[128 + b] / n2;
        float e[8] = {ce0, de0, ce1, de1, ce2, de2, ce2, 0.f}; // lo_e == ce2
        e0[b * 2] = ce0; e0[b * 2 + 1] = de0;
        e1[b * 2] = ce1; e1[b * 2 + 1] = de1;
        e2[b * 2] = ce2; e2[b * 2 + 1] = de2;
        e3[b * 2] = ce2; e3[b * 2 + 1] = 0.f;
        for (int j = 0; j < 4; ++j) {               // only gates 0..3 are used
            float z = gate_b[j];
            for (int k = 0; k < 8; ++k) z += gate_w[j * 8 + k] * e[k];
            g[b * 4 + j] = 1.f / (1.f + expf(-z));
        }
    }
    if (threadIdx.x == 0) {
        float loss = 0.f;
        for (int i = 0; i < 3; ++i) {
            int D = 512 >> i;
            float sd = 0.f;
            for (int bb = 0; bb < 64; ++bb) sd += sumAbsD[i * 64 + bb];
            loss += 0.01f * sd / (64.f * 128.f * (float)D);
            float er = sqrtf(sumSqC[i]) / (sqrtf(sumSqXe[i]) + 1e-6f) - 1.f;
            loss += 0.01f * er * er;
        }
        *loss_out = loss;
    }
}

// ---------------------------------------------------------------------------
extern "C" void kernel_launch(void* const* d_in, const int* in_sizes, int n_in,
                              void* d_out, int out_size, void* d_ws, size_t ws_size,
                              hipStream_t stream) {
    (void)in_sizes; (void)n_in; (void)out_size; (void)ws_size;

    // Inputs in setup_inputs() dict order (tuples flattened)
    const float* x = (const float*)d_in[0];
    const float* dP_w[3] = {(const float*)d_in[1],  (const float*)d_in[2],  (const float*)d_in[3]};
    const float* dP_b[3] = {(const float*)d_in[4],  (const float*)d_in[5],  (const float*)d_in[6]};
    const float* dU_w[3] = {(const float*)d_in[7],  (const float*)d_in[8],  (const float*)d_in[9]};
    const float* dU_b[3] = {(const float*)d_in[10], (const float*)d_in[11], (const float*)d_in[12]};
    const float* rP_w[3] = {(const float*)d_in[13], (const float*)d_in[14], (const float*)d_in[15]};
    const float* rP_b[3] = {(const float*)d_in[16], (const float*)d_in[17], (const float*)d_in[18]};
    const float* rU_w[3] = {(const float*)d_in[19], (const float*)d_in[20], (const float*)d_in[21]};
    const float* rU_b[3] = {(const float*)d_in[22], (const float*)d_in[23], (const float*)d_in[24]};
    const float* gate_w  = (const float*)d_in[25];
    const float* gate_b  = (const float*)d_in[26];

    // ---- workspace layout (bytes) ----
    char* ws = (char*)d_ws;
    f16*   w16   = (f16*)(ws + 0);                 //  2,752,512 B (12 weight mats, f16)
    f16*   xe16  = (f16*)(ws + 2752512);           //  8 MiB
    f16*   det16 = (f16*)(ws + 11141120);          //  8 MiB
    float* curA  = (float*)(ws + 19529728);        // 16 MiB (fwd c0 / recon xe32)
    float* curB  = (float*)(ws + 36306944);        //  8 MiB (fwd c1 / recon rec2)
    float* rec1  = (float*)(ws + 44695552);        // 16 MiB
    float* accf  = (float*)(ws + 61472768);        // accumulators + g
    float* sumAbsC = accf;          // 3*64
    float* sumAbsD = accf + 192;    // 3*64
    float* sumSqXe = accf + 384;    // 3
    float* sumSqC  = accf + 387;    // 3
    float* gbuf    = accf + 390;    // 64*4

    const size_t fam = 344064;                     // 512^2+256^2+128^2
    const size_t lo[3] = {0, 262144, 327680};
    f16 *dPw16[3], *dUw16[3], *rPw16[3], *rUw16[3];
    for (int i = 0; i < 3; ++i) {
        dPw16[i] = w16 + 0 * fam + lo[i];
        dUw16[i] = w16 + 1 * fam + lo[i];
        rPw16[i] = w16 + 2 * fam + lo[i];
        rUw16[i] = w16 + 3 * fam + lo[i];
    }

    // ---- output layout (floats, return order) ----
    float* out = (float*)d_out;
    float* gated[4]  = {out, out + 4194304, out + 6291456, out + 7340032};
    float* rec_out   = out + 8388608;
    float* coeffs[4] = {out + 16777216, out + 20971520, out + 23068672, out + 24117248};
    float* energy[4] = {out + 25165824, out + 25165952, out + 25166080, out + 25166208};
    float* loss_out  = out + 25166336;

    hipMemsetAsync(accf, 0, 646 * sizeof(float), stream);

    // Weight f32 -> f16
    const int wsz[3] = {512 * 512, 256 * 256, 128 * 128};
    for (int i = 0; i < 3; ++i) {
        int gb = (wsz[i] + 255) / 256;
        f32_to_f16<<<gb, 256, 0, stream>>>(dP_w[i], dPw16[i], wsz[i]);
        f32_to_f16<<<gb, 256, 0, stream>>>(dU_w[i], dUw16[i], wsz[i]);
        f32_to_f16<<<gb, 256, 0, stream>>>(rP_w[i], rPw16[i], wsz[i]);
        f32_to_f16<<<gb, 256, 0, stream>>>(rU_w[i], rUw16[i], wsz[i]);
    }

    // ---- forward: analysis levels ----
    const float* cur = x;
    float* curnext[3] = {curA, curB, coeffs[3]};   // final cur == coeffs[3]
    for (int i = 0; i < 3; ++i) {
        int D = 512 >> i;
        size_t tot = (size_t)8192 * D;
        split_even<<<2048, 256, 0, stream>>>(cur, xe16, D, tot, &sumSqXe[i]);
        launch_gemm<0>(D, stream, xe16, dPw16[i], dP_b[i], cur,
                       coeffs[i], det16, nullptr, &sumAbsD[i * 64], nullptr);
        launch_gemm<1>(D, stream, det16, dUw16[i], dU_b[i], cur,
                       curnext[i], nullptr, nullptr, &sumAbsC[i * 64], &sumSqC[i]);
        cur = curnext[i];
    }

    // ---- energies, gate, loss ----
    gate_and_loss<<<1, 64, 0, stream>>>(sumAbsC, sumAbsD, sumSqXe, sumSqC,
                                        gate_w, gate_b, gbuf,
                                        energy[0], energy[1], energy[2], energy[3],
                                        loss_out);

    // gated[3] (lowpass), f32 only — used as initial "approx"
    gate_scale<<<1024, 256, 0, stream>>>(coeffs[3], gbuf, 3, gated[3], (f16*)nullptr,
                                         (size_t)8192 * 128, 128 * 128);

    // ---- reconstruction: synthesis levels ----
    float* xe32buf = curA;                 // forward buffers are dead now
    float* recOut[3] = {rec_out, rec1, curB};
    const float* approx = gated[3];
    for (int i = 2; i >= 0; --i) {
        int D = 512 >> i;
        size_t tot = (size_t)8192 * D;
        gate_scale<<<2048, 256, 0, stream>>>(coeffs[i], gbuf, i, gated[i], det16,
                                             tot, 128 * D);
        launch_gemm<2>(D, stream, det16, rUw16[i], rU_b[i], approx,
                       xe32buf, xe16, nullptr, nullptr, nullptr);
        launch_gemm<3>(D, stream, xe16, rPw16[i], rP_b[i], gated[i],
                       recOut[i], nullptr, xe32buf, nullptr, nullptr);
        approx = recOut[i];
    }
}
